// FixedPointSolver_77756087927332
// MI455X (gfx1250) — matrix operand order repaired and checked by reading, not compiled, not run on hardware
//
#include <hip/hip_runtime.h>
#include <math.h>

// ---------------------------------------------------------------------------
// Problem constants (from reference)
// ---------------------------------------------------------------------------
#define B_   4
#define S_   512
#define D_   1024
#define H_   16
#define DH_  64
#define DQ_  256
#define MTOK (B_ * S_)          // 2048 tokens
#define MAXST 12
#define MINST 4
#define THR_ 1e-4f

// CDNA5 WMMA vector types (probe-confirmed signatures)
typedef __attribute__((ext_vector_type(16))) __bf16 v16bf;
typedef __attribute__((ext_vector_type(8)))  float  v8f;
typedef __attribute__((ext_vector_type(4)))  unsigned int u32x4;
typedef __attribute__((ext_vector_type(8)))  int i32x8;
typedef __attribute__((ext_vector_type(4)))  int i32x4;

// Tensor Data Mover availability (device pass only; host pass falls back)
#if defined(__gfx1250__) && __has_builtin(__builtin_amdgcn_tensor_load_to_lds)
#define TDM_OK 1
#else
#define TDM_OK 0
#endif

// ---------------------------------------------------------------------------
// Helpers
// ---------------------------------------------------------------------------
__device__ __forceinline__ unsigned short f2bf(float f) {
  unsigned int u = __float_as_uint(f);
  u += 0x7FFFu + ((u >> 16) & 1u);          // round-to-nearest-even
  return (unsigned short)(u >> 16);
}
__device__ __forceinline__ float gelu_f(float x) {
  return 0.5f * x * (1.0f + erff(x * 0.70710678118654752f));   // exact GELU
}
__device__ __forceinline__ float sigm_f(float x) {
  return 1.0f / (1.0f + expf(-x));
}

// Epilogue variants
enum {
  EPI_F32_SCALE = 0,       // C.f32 = acc * cscale              (attention scores)
  EPI_BIAS_GELU_BF16 = 1,  // C.bf16 = gelu(acc + bias)         (MLP hidden)
  EPI_BIAS_BF16 = 2,       // C.bf16 = acc + bias               (Q/K/V)
  EPI_BLEND_BF16 = 3,      // g=sig(acc+bias); C.bf16=g*h+(1-g)*x
  EPI_G1_F32 = 4,          // C.f32 = sig(acc+bias) * (*scalep) (G1 gate)
  EPI_BIAS_F32 = 5,        // C.f32 = acc + bias                (phi)
  EPI_BF16 = 6             // C.bf16 = acc                      (attention ctx)
};

// ---------------------------------------------------------------------------
// bf16 WMMA GEMM:  C[M,N] = A[M,K] * op(B) (+ epilogue)
//   BT=1: B is [N,K] row-major (weights / K-matrix) -> "NT" GEMM
//         B tile staged via Tensor Data Mover (wave 0, TENSORcnt) when avail.
//   BT=0: B is [K,N] row-major (V-matrix)           -> "NN" GEMM
// 256 threads = 8 wave32 waves in 4(M) x 2(N); workgroup tile 128x64,
// per-wave 32x32 = 2x2 tiles of v_wmma_f32_16x16x32_bf16; K step 32.
// Batched via blockIdx.z with (z/zdiv, z%zdiv) stride pairs.
// ---------------------------------------------------------------------------
template<int BT, int EPI>
__global__ __launch_bounds__(256)
void gemm_bf16_wmma(const unsigned short* __restrict__ A,
                    const unsigned short* __restrict__ Bm,
                    void* __restrict__ Cv,
                    const float* __restrict__ bias,
                    const float* __restrict__ aux0,
                    const float* __restrict__ aux1,
                    const float* __restrict__ scalep,
                    int M, int N, int K, int lda, int ldb, int ldc,
                    long sA0, long sA1, long sB0, long sB1, long sC0, long sC1,
                    int zdiv, float cscale)
{
  __shared__ __align__(16) unsigned short lA[128 * 32];   // [m][k]
  // BT=1: [n][k] (TDM writes X-contiguous rows);  BT=0: [k][n]
  __shared__ __align__(16) unsigned short lB[64 * 32];

  const int tid    = threadIdx.x;
  const int lane   = tid & 31;
  const int wave   = tid >> 5;
  const int wm     = wave & 3;
  const int wn     = wave >> 2;
  const int laneHi = lane >> 4;
  const int lane16 = lane & 15;

  const int bm0 = blockIdx.x * 128;
  const int bn0 = blockIdx.y * 64;
  const int z   = blockIdx.z;
  const long zb = z / zdiv, zh = z % zdiv;

  const unsigned short* Ag = A  + zb * sA0 + zh * sA1;
  const unsigned short* Bg = Bm + zb * sB0 + zh * sB1;
  const long cOff = zb * sC0 + zh * sC1;

  v8f acc[2][2] = {};

  for (int k0 = 0; k0 < K; k0 += 32) {
#if TDM_OK
    const bool tdmB = (BT == 1);
    if (tdmB && wave == 0) {
      // --- Tensor DMA: stage 64(n) x 32(k) bf16 B tile into lB ------------
      // D# group0: count=1 | lds_addr | global_addr | type=2
      // (LDS byte address: workgroup-relative offset of lB)
      unsigned ldsB = (unsigned)(unsigned long long)&lB[0];
      unsigned long long ga =
          (unsigned long long)(const void*)(Bg + (size_t)bn0 * ldb + k0);
      u32x4 g0;
      g0.x = 1u;                                    // count=1, user mode
      g0.y = ldsB;                                  // lds_addr (bytes)
      g0.z = (unsigned)(ga & 0xffffffffu);          // global_addr[31:0]
      g0.w = (unsigned)((ga >> 32) & 0x01ffffffu)   // global_addr[56:32]
             | (2u << 30);                          // type=2 ("image")
      // D# group1: data_size=2B, tensor K x N, tile 32 x 64, stride=ldb
      unsigned td0 = (unsigned)K, td1 = (unsigned)N;
      unsigned long long st0 = (unsigned long long)ldb;
      i32x8 g1;
      g1[0] = (int)(1u << 16);                              // data_size=1 (2B)
      g1[1] = (int)((td0 & 0xffffu) << 16);                 // tensor_dim0 lo
      g1[2] = (int)((td0 >> 16) | ((td1 & 0xffffu) << 16)); // dim0 hi | dim1 lo
      g1[3] = (int)((td1 >> 16) | (32u << 16));             // dim1 hi | tile_dim0=32
      g1[4] = (int)(64u);                                   // tile_dim1=64, tile_dim2=0
      g1[5] = (int)(st0 & 0xffffffffu);                     // tensor_dim0_stride lo
      g1[6] = (int)((st0 >> 32) & 0xffffu);                 // stride hi | dim1_stride lo
      g1[7] = 0;
      i32x4 gz = {0, 0, 0, 0};
#if __has_include(<hip/amd_detail/amd_gfx1250_TDM.h>)
      i32x8 gz8 = {0, 0, 0, 0, 0, 0, 0, 0};
      __builtin_amdgcn_tensor_load_to_lds(g0, g1, gz, gz, gz8, 0);
#else
      __builtin_amdgcn_tensor_load_to_lds(g0, g1, gz, gz, 0);
#endif
    }
#else
    const bool tdmB = false;
#endif

    // --- stage A tile 128x32 via b128 loads buffered in registers ---------
    u32x4 va[2];
#pragma unroll
    for (int i = 0; i < 2; ++i) {
      int e = tid + i * 256;             // 512 x b128 chunks
      int m = e >> 2, kq = (e & 3) * 8;
      va[i] = *(const u32x4*)(Ag + (size_t)(bm0 + m) * lda + k0 + kq);
    }
    u32x4 vbuf = {};
    if (!tdmB) {
      if (BT) {                          // B[N,K] -> lB[n][k]
        int n = tid >> 2, kq = (tid & 3) * 8;
        vbuf = *(const u32x4*)(Bg + (size_t)(bn0 + n) * ldb + k0 + kq);
      } else {                           // B[K,N] -> lB[k][n]
        int k = tid >> 3, nq = (tid & 7) * 8;
        vbuf = *(const u32x4*)(Bg + (size_t)(k0 + k) * ldb + bn0 + nq);
      }
    }
#pragma unroll
    for (int i = 0; i < 2; ++i) {
      int e = tid + i * 256;
      int m = e >> 2, kq = (e & 3) * 8;
      *(u32x4*)(&lA[m * 32 + kq]) = va[i];
    }
    if (!tdmB) {
      if (BT) { int n = tid >> 2, kq = (tid & 3) * 8; *(u32x4*)(&lB[n * 32 + kq]) = vbuf; }
      else    { int k = tid >> 3, nq = (tid & 7) * 8; *(u32x4*)(&lB[k * 64 + nq]) = vbuf; }
    }
    if (k0 + 32 < K) {                   // stream next K tile (global_prefetch_b8)
      __builtin_prefetch(Ag + (size_t)(bm0 + (tid >> 1)) * lda + k0 + 32, 0, 1);
      if (BT && !tdmB)
        __builtin_prefetch(Bg + (size_t)(bn0 + (tid & 63)) * ldb + k0 + 32, 0, 1);
    }
#if TDM_OK
    if (tdmB && wave == 0) __builtin_amdgcn_s_wait_tensorcnt(0);
#endif
    __syncthreads();

    union Frag { v16bf v; unsigned int u[8]; };
    Frag a[2], b[2];
    // A fragment per ISA 16-bit A layout (lanes 0-15: M, VGPRs: K pairs)
#pragma unroll
    for (int ti = 0; ti < 2; ++ti) {
      int m = wm * 32 + ti * 16 + lane16;
#pragma unroll
      for (int v = 0; v < 8; ++v) {
        int k = ((v < 4) ? (v * 2) : (16 + (v - 4) * 2)) + (laneHi ? 8 : 0);
        a[ti].u[v] = *(const unsigned int*)(&lA[m * 32 + k]);
      }
    }
    // B fragment: same K order as A (lanes 0-15: K 0-7,16-23; lanes 16-31: K 8-15,24-31)
#pragma unroll
    for (int tj = 0; tj < 2; ++tj) {
      int n = wn * 32 + tj * 16 + lane16;
      int koff = laneHi ? 8 : 0;
#pragma unroll
      for (int v = 0; v < 8; ++v) {
        int k = koff + ((v < 4) ? (v * 2) : (16 + (v - 4) * 2));
        if (BT) {
          b[tj].u[v] = *(const unsigned int*)(&lB[n * 32 + k]);
        } else {
          unsigned lo = lB[k * 64 + n];
          unsigned hi = lB[(k + 1) * 64 + n];
          b[tj].u[v] = lo | (hi << 16);
        }
      }
    }
#pragma unroll
    for (int ti = 0; ti < 2; ++ti)
#pragma unroll
      for (int tj = 0; tj < 2; ++tj)
        acc[ti][tj] = __builtin_amdgcn_wmma_f32_16x16x32_bf16(
            false, a[ti].v, false, b[tj].v, (short)0, acc[ti][tj], false, false);
    __syncthreads();
  }

  // epilogue: D layout -> VGPR r: m = r + 8*laneHi, n = lane%16
  float*          Cf = (float*)Cv;
  unsigned short* Cu = (unsigned short*)Cv;
#pragma unroll
  for (int ti = 0; ti < 2; ++ti) {
#pragma unroll
    for (int tj = 0; tj < 2; ++tj) {
#pragma unroll
      for (int r = 0; r < 8; ++r) {
        int m = bm0 + wm * 32 + ti * 16 + r + laneHi * 8;
        int n = bn0 + wn * 32 + tj * 16 + lane16;
        size_t ci = (size_t)cOff + (size_t)m * ldc + n;
        float v = acc[ti][tj][r];
        if (EPI == EPI_F32_SCALE) {
          Cf[ci] = v * cscale;
        } else if (EPI == EPI_BIAS_GELU_BF16) {
          Cu[ci] = f2bf(gelu_f(v + bias[n]));
        } else if (EPI == EPI_BIAS_BF16) {
          Cu[ci] = f2bf(v + bias[n]);
        } else if (EPI == EPI_BLEND_BF16) {
          float g = sigm_f(v + bias[n]);
          size_t ai = (size_t)m * ldc + n;
          Cu[ci] = f2bf(g * aux0[ai] + (1.0f - g) * aux1[ai]);
        } else if (EPI == EPI_G1_F32) {
          Cf[ci] = sigm_f(v + bias[n]) * scalep[0];
        } else if (EPI == EPI_BIAS_F32) {
          Cf[ci] = v + bias[n];
        } else { // EPI_BF16
          Cu[ci] = f2bf(v);
        }
      }
    }
  }
}

// ---------------------------------------------------------------------------
// Small support kernels
// ---------------------------------------------------------------------------
__global__ void cvt_bf16_kernel(unsigned short* __restrict__ o,
                                const float* __restrict__ a, int n) {
  int i = blockIdx.x * blockDim.x + threadIdx.x;
  if (i < n) o[i] = f2bf(a[i]);
}

__global__ void concat_cvt_kernel(unsigned short* __restrict__ hx,
                                  unsigned short* __restrict__ hb,
                                  const float* __restrict__ h,
                                  const float* __restrict__ x, int n) {
  int i = blockIdx.x * blockDim.x + threadIdx.x;
  if (i >= n) return;
  int m = i >> 10, d = i & 1023;
  unsigned short hv = f2bf(h[i]);
  hb[i] = hv;
  hx[(size_t)m * 2048 + d]        = hv;
  hx[(size_t)m * 2048 + 1024 + d] = f2bf(x[i]);
}

__global__ void pool_kernel(float* __restrict__ pooled, const float* __restrict__ h) {
  int i = blockIdx.x * blockDim.x + threadIdx.x;     // 0..B*D-1
  if (i >= B_ * D_) return;
  int b = i >> 10, d = i & 1023;
  float s = 0.f;
  for (int t = 0; t < S_; ++t) s += h[((size_t)b * S_ + t) * D_ + d];
  pooled[i] = s * (1.0f / S_);
}

// ctrl layout: [0]=gw0 [1]=res_num [2]=res_den [3]=num_steps [4+i]=active_i
__global__ __launch_bounds__(256)
void predictor_kernel(float* __restrict__ ctrl, const float* __restrict__ pooled,
                      const float* __restrict__ w1, const float* __restrict__ b1,
                      const float* __restrict__ w2, const float* __restrict__ b2,
                      const float* __restrict__ gl) {
  __shared__ float red[256];
  __shared__ float outb[B_];
  int j = threadIdx.x;
  float t[B_];
  for (int b = 0; b < B_; ++b) {
    float acc = b1[j];
    const float* wr = w1 + (size_t)j * D_;
    const float* pr = pooled + (size_t)b * D_;
    for (int d = 0; d < D_; ++d) acc += pr[d] * wr[d];
    t[b] = gelu_f(acc);
  }
  float w2j = w2[j];
  for (int b = 0; b < B_; ++b) {
    red[j] = t[b] * w2j;
    __syncthreads();
    for (int s = 128; s > 0; s >>= 1) { if (j < s) red[j] += red[j + s]; __syncthreads(); }
    if (j == 0) outb[b] = red[0] + b2[0];
    __syncthreads();
  }
  if (j == 0) {
    float ex = 0.f;
    for (int b = 0; b < B_; ++b) ex += sigm_f(outb[b]);
    ex *= 0.25f;
    int ns = MINST + (int)floorf(ex * (float)(MAXST - MINST));
    if (ns > MAXST) ns = MAXST;
    float m  = fmaxf(gl[0], fmaxf(gl[1], gl[2]));
    float e0 = expf(gl[0] - m), e1 = expf(gl[1] - m), e2 = expf(gl[2] - m);
    ctrl[0] = e0 / (e0 + e1 + e2);        // softmax(gate_logits)[0]
    ctrl[1] = 0.f; ctrl[2] = 0.f;
    ctrl[3] = (float)ns;
    ctrl[4] = 1.f;
    for (int i = 1; i <= MAXST; ++i) ctrl[4 + i] = 0.f;
  }
}

__global__ void zero_res_kernel(float* ctrl) {
  if (threadIdx.x == 0) { ctrl[1] = 0.f; ctrl[2] = 0.f; }
}

// wave32 per row: 512-wide softmax with shuffle reductions, writes bf16 P
__global__ __launch_bounds__(256)
void softmax_kernel(unsigned short* __restrict__ P, const float* __restrict__ Sc, int rows) {
  int wave = threadIdx.x >> 5, lane = threadIdx.x & 31;
  int row = blockIdx.x * 8 + wave;
  if (row >= rows) return;
  const float* r = Sc + (size_t)row * S_;
  float v[16];
  float mx = -3.4e38f;
#pragma unroll
  for (int t = 0; t < 16; ++t) { v[t] = r[lane + t * 32]; mx = fmaxf(mx, v[t]); }
#pragma unroll
  for (int s = 16; s > 0; s >>= 1) mx = fmaxf(mx, __shfl_xor(mx, s, 32));
  float sum = 0.f;
#pragma unroll
  for (int t = 0; t < 16; ++t) { v[t] = expf(v[t] - mx); sum += v[t]; }
#pragma unroll
  for (int s = 16; s > 0; s >>= 1) sum += __shfl_xor(sum, s, 32);
  float inv = 1.0f / sum;
  unsigned short* o = P + (size_t)row * S_;
#pragma unroll
  for (int t = 0; t < 16; ++t) o[lane + t * 32] = f2bf(v[t] * inv);
}

// LayerNorm(h + G1*phi) per token + accumulate ||hn-h||^2 and ||h||^2
__global__ __launch_bounds__(256)
void ln_stats_kernel(float* __restrict__ hn, float* __restrict__ ctrl,
                     const float* __restrict__ h, const float* __restrict__ G1,
                     const float* __restrict__ phi,
                     const float* __restrict__ g, const float* __restrict__ bb) {
  __shared__ float red[256];
  int m = blockIdx.x, tid = threadIdx.x;
  size_t base = (size_t)m * D_;
  float y[4], hv[4];
  float s = 0.f;
#pragma unroll
  for (int k = 0; k < 4; ++k) {
    int d = tid + k * 256;
    hv[k] = h[base + d];
    y[k]  = hv[k] + G1[base + d] * phi[base + d];
    s += y[k];
  }
  red[tid] = s; __syncthreads();
  for (int t = 128; t > 0; t >>= 1) { if (tid < t) red[tid] += red[tid + t]; __syncthreads(); }
  float mean = red[0] * (1.0f / D_);
  __syncthreads();
  float s2 = 0.f;
#pragma unroll
  for (int k = 0; k < 4; ++k) { float dl = y[k] - mean; s2 += dl * dl; }
  red[tid] = s2; __syncthreads();
  for (int t = 128; t > 0; t >>= 1) { if (tid < t) red[tid] += red[tid + t]; __syncthreads(); }
  float rstd = rsqrtf(red[0] * (1.0f / D_) + 1e-5f);
  __syncthreads();
  float dsq = 0.f, hsq = 0.f;
#pragma unroll
  for (int k = 0; k < 4; ++k) {
    int d = tid + k * 256;
    float o = (y[k] - mean) * rstd * g[d] + bb[d];
    hn[base + d] = o;
    float df = o - hv[k];
    dsq += df * df; hsq += hv[k] * hv[k];
  }
  red[tid] = dsq; __syncthreads();
  for (int t = 128; t > 0; t >>= 1) { if (tid < t) red[tid] += red[tid + t]; __syncthreads(); }
  if (tid == 0) atomicAdd(&ctrl[1], red[0]);
  __syncthreads();
  red[tid] = hsq; __syncthreads();
  for (int t = 128; t > 0; t >>= 1) { if (tid < t) red[tid] += red[tid + t]; __syncthreads(); }
  if (tid == 0) atomicAdd(&ctrl[2], red[0]);
}

// Device-side predicated commit: h = act ? hn : h; chain active flag
__global__ void commit_kernel(float* __restrict__ h, const float* __restrict__ hn,
                              float* __restrict__ ctrl, int step, int n) {
  int i = blockIdx.x * blockDim.x + threadIdx.x;
  bool act = (ctrl[4 + step] != 0.f) && (step < (int)ctrl[3]);
  if (i < n && act) h[i] = hn[i];
  if (i == 0) {
    float res = sqrtf(ctrl[1]) / (sqrtf(ctrl[2]) + 1e-8f);
    bool nxt = act && !((step >= MINST) && (res < THR_));
    ctrl[4 + step + 1] = nxt ? 1.f : 0.f;
  }
}

// ---------------------------------------------------------------------------
// Host orchestration
// ---------------------------------------------------------------------------
extern "C" void kernel_launch(void* const* d_in, const int* in_sizes, int n_in,
                              void* d_out, int out_size, void* d_ws, size_t ws_size,
                              hipStream_t stream) {
  const float* h_in  = (const float*)d_in[0];
  const float* x     = (const float*)d_in[1];
  const float* ig_w1 = (const float*)d_in[2];  const float* ig_b1 = (const float*)d_in[3];
  const float* ig_w2 = (const float*)d_in[4];  const float* ig_b2 = (const float*)d_in[5];
  const float* wq = (const float*)d_in[6];  const float* bq = (const float*)d_in[7];
  const float* wk = (const float*)d_in[8];  const float* bk = (const float*)d_in[9];
  const float* wv = (const float*)d_in[10]; const float* bv = (const float*)d_in[11];
  const float* wo = (const float*)d_in[12]; const float* bo = (const float*)d_in[13];
  const float* g1_w1 = (const float*)d_in[14]; const float* g1_b1 = (const float*)d_in[15];
  const float* g1_w2 = (const float*)d_in[16]; const float* g1_b2 = (const float*)d_in[17];
  const float* gl = (const float*)d_in[18];
  const float* ln_g = (const float*)d_in[19]; const float* ln_b = (const float*)d_in[20];
  const float* sp_w1 = (const float*)d_in[21]; const float* sp_b1 = (const float*)d_in[22];
  const float* sp_w2 = (const float*)d_in[23]; const float* sp_b2 = (const float*)d_in[24];

  char* p = (char*)d_ws;
  auto alloc = [&](size_t bytes) -> void* {
    void* r = (void*)p;
    p += (bytes + 255) & ~(size_t)255;
    return r;
  };

  float* h_cur  = (float*)alloc((size_t)MTOK * D_ * 4);
  float* hn     = (float*)alloc((size_t)MTOK * D_ * 4);
  float* G1     = (float*)alloc((size_t)MTOK * D_ * 4);
  float* phi    = (float*)alloc((size_t)MTOK * D_ * 4);
  float* pooled = (float*)alloc((size_t)B_ * D_ * 4);
  float* ctrl   = (float*)alloc(64 * 4);
  unsigned short* w_ig1 = (unsigned short*)alloc((size_t)DQ_ * 2 * D_ * 2);
  unsigned short* w_ig2 = (unsigned short*)alloc((size_t)D_ * DQ_ * 2);
  unsigned short* w_g1a = (unsigned short*)alloc((size_t)DQ_ * D_ * 2);
  unsigned short* w_g1b = (unsigned short*)alloc((size_t)D_ * DQ_ * 2);
  unsigned short* w_q   = (unsigned short*)alloc((size_t)D_ * D_ * 2);
  unsigned short* w_k   = (unsigned short*)alloc((size_t)D_ * D_ * 2);
  unsigned short* w_v   = (unsigned short*)alloc((size_t)D_ * D_ * 2);
  unsigned short* w_o   = (unsigned short*)alloc((size_t)D_ * D_ * 2);
  unsigned short* hx    = (unsigned short*)alloc((size_t)MTOK * 2 * D_ * 2);
  unsigned short* hb    = (unsigned short*)alloc((size_t)MTOK * D_ * 2);
  unsigned short* t1    = (unsigned short*)alloc((size_t)MTOK * DQ_ * 2);
  unsigned short* u1    = (unsigned short*)alloc((size_t)MTOK * DQ_ * 2);
  unsigned short* blend = (unsigned short*)alloc((size_t)MTOK * D_ * 2);
  unsigned short* qb    = (unsigned short*)alloc((size_t)MTOK * D_ * 2);
  unsigned short* kb    = (unsigned short*)alloc((size_t)MTOK * D_ * 2);
  unsigned short* vb    = (unsigned short*)alloc((size_t)MTOK * D_ * 2);
  float*          sc    = (float*)alloc((size_t)B_ * H_ * S_ * S_ * 4);
  unsigned short* Pm    = (unsigned short*)alloc((size_t)B_ * H_ * S_ * S_ * 2);
  unsigned short* ctx   = (unsigned short*)alloc((size_t)MTOK * D_ * 2);

  hipMemcpyAsync(h_cur, h_in, (size_t)MTOK * D_ * 4, hipMemcpyDeviceToDevice, stream);

  auto cvt = [&](unsigned short* dst, const float* src, int n) {
    cvt_bf16_kernel<<<(n + 255) / 256, 256, 0, stream>>>(dst, src, n);
  };
  cvt(w_ig1, ig_w1, DQ_ * 2 * D_);
  cvt(w_ig2, ig_w2, D_ * DQ_);
  cvt(w_g1a, g1_w1, DQ_ * D_);
  cvt(w_g1b, g1_w2, D_ * DQ_);
  cvt(w_q, wq, D_ * D_); cvt(w_k, wk, D_ * D_);
  cvt(w_v, wv, D_ * D_); cvt(w_o, wo, D_ * D_);

  pool_kernel<<<(B_ * D_ + 255) / 256, 256, 0, stream>>>(pooled, h_in);
  predictor_kernel<<<1, 256, 0, stream>>>(ctrl, pooled, sp_w1, sp_b1, sp_w2, sp_b2, gl);

  const dim3 blk(256);
  for (int step = 0; step < MAXST; ++step) {
    zero_res_kernel<<<1, 32, 0, stream>>>(ctrl);
    concat_cvt_kernel<<<(MTOK * D_ + 255) / 256, 256, 0, stream>>>(hx, hb, h_cur, x, MTOK * D_);

    // t1 = gelu(concat(h,x) @ ig_w1^T + b1)   M=2048 N=256 K=2048
    gemm_bf16_wmma<1, EPI_BIAS_GELU_BF16><<<dim3(MTOK / 128, DQ_ / 64, 1), blk, 0, stream>>>(
        hx, w_ig1, t1, ig_b1, nullptr, nullptr, nullptr,
        MTOK, DQ_, 2 * D_, 2 * D_, 2 * D_, DQ_, 0, 0, 0, 0, 0, 0, 1, 1.f);
    // blended = sig(t1 @ ig_w2^T + b2)*h + (1-g)*x   M=2048 N=1024 K=256
    gemm_bf16_wmma<1, EPI_BLEND_BF16><<<dim3(MTOK / 128, D_ / 64, 1), blk, 0, stream>>>(
        t1, w_ig2, blend, ig_b2, h_cur, x, nullptr,
        MTOK, D_, DQ_, DQ_, DQ_, D_, 0, 0, 0, 0, 0, 0, 1, 1.f);
    // u1 = gelu(h @ g1_w1^T + b1)   M=2048 N=256 K=1024
    gemm_bf16_wmma<1, EPI_BIAS_GELU_BF16><<<dim3(MTOK / 128, DQ_ / 64, 1), blk, 0, stream>>>(
        hb, w_g1a, u1, g1_b1, nullptr, nullptr, nullptr,
        MTOK, DQ_, D_, D_, D_, DQ_, 0, 0, 0, 0, 0, 0, 1, 1.f);
    // G1 = sig(u1 @ g1_w2^T + b2) * gw0   M=2048 N=1024 K=256
    gemm_bf16_wmma<1, EPI_G1_F32><<<dim3(MTOK / 128, D_ / 64, 1), blk, 0, stream>>>(
        u1, w_g1b, G1, g1_b2, nullptr, nullptr, ctrl,
        MTOK, D_, DQ_, DQ_, DQ_, D_, 0, 0, 0, 0, 0, 0, 1, 1.f);
    // Q / K / V   M=2048 N=1024 K=1024
    gemm_bf16_wmma<1, EPI_BIAS_BF16><<<dim3(MTOK / 128, D_ / 64, 1), blk, 0, stream>>>(
        blend, w_q, qb, bq, nullptr, nullptr, nullptr,
        MTOK, D_, D_, D_, D_, D_, 0, 0, 0, 0, 0, 0, 1, 1.f);
    gemm_bf16_wmma<1, EPI_BIAS_BF16><<<dim3(MTOK / 128, D_ / 64, 1), blk, 0, stream>>>(
        blend, w_k, kb, bk, nullptr, nullptr, nullptr,
        MTOK, D_, D_, D_, D_, D_, 0, 0, 0, 0, 0, 0, 1, 1.f);
    gemm_bf16_wmma<1, EPI_BIAS_BF16><<<dim3(MTOK / 128, D_ / 64, 1), blk, 0, stream>>>(
        blend, w_v, vb, bv, nullptr, nullptr, nullptr,
        MTOK, D_, D_, D_, D_, D_, 0, 0, 0, 0, 0, 0, 1, 1.f);
    // scores = (Q @ K^T) * 1/sqrt(64), batched over (b,h)=64
    gemm_bf16_wmma<1, EPI_F32_SCALE><<<dim3(S_ / 128, S_ / 64, B_ * H_), blk, 0, stream>>>(
        qb, kb, sc, nullptr, nullptr, nullptr, nullptr,
        S_, S_, DH_, D_, D_, S_,
        (long)S_ * D_, 64, (long)S_ * D_, 64, (long)H_ * S_ * S_, (long)S_ * S_, H_, 0.125f);
    softmax_kernel<<<(B_ * H_ * S_) / 8, 256, 0, stream>>>(Pm, sc, B_ * H_ * S_);
    // ctx = P @ V (NN), batched
    gemm_bf16_wmma<0, EPI_BF16><<<dim3(S_ / 128, DH_ / 64, B_ * H_), blk, 0, stream>>>(
        Pm, vb, ctx, nullptr, nullptr, nullptr, nullptr,
        S_, DH_, S_, S_, D_, D_,
        (long)H_ * S_ * S_, (long)S_ * S_, (long)S_ * D_, 64, (long)S_ * D_, 64, H_, 1.f);
    // phi = ctx @ wo^T + bo   M=2048 N=1024 K=1024
    gemm_bf16_wmma<1, EPI_BIAS_F32><<<dim3(MTOK / 128, D_ / 64, 1), blk, 0, stream>>>(
        ctx, w_o, phi, bo, nullptr, nullptr, nullptr,
        MTOK, D_, D_, D_, D_, D_, 0, 0, 0, 0, 0, 0, 1, 1.f);

    ln_stats_kernel<<<MTOK, 256, 0, stream>>>(hn, ctrl, h_cur, G1, phi, ln_g, ln_b);
    commit_kernel<<<(MTOK * D_ + 255) / 256, 256, 0, stream>>>(h_cur, hn, ctrl, step, MTOK * D_);
  }

  hipMemcpyAsync(d_out, h_cur, (size_t)MTOK * D_ * 4, hipMemcpyDeviceToDevice, stream);
}
